// HardNegativeWeightedFConLoss_63866163691792
// MI455X (gfx1250) — matrix operand order repaired and checked by reading, not compiled, not run on hardware
//
#include <hip/hip_runtime.h>

// ---------------------------------------------------------------------------
// HardNegativeWeightedFConLoss for MI455X (gfx1250, wave32, WMMA bf16)
//
// Heavy GEMMs (adj@emb x2, Fn@Mn^T x2, feats@W1) run on v_wmma_f32_16x16x32_bf16
// with f32 accumulation. Operand staging uses CDNA5 async Global->LDS loads
// (global_load_async_to_lds_b128 + s_wait_asynccnt) with double-buffered LDS
// tiles for load/compute overlap — no VGPR round trip, no s_wait_loadcnt stalls.
// ---------------------------------------------------------------------------

typedef __attribute__((ext_vector_type(16))) __bf16 v16bf;
typedef __attribute__((ext_vector_type(8)))  float  v8f;

static constexpr int NROW = 4096;   // N
static constexpr int DIM  = 512;    // d
static constexpr int DIM2 = 1024;   // 2d
static constexpr int TPB  = 256;    // 8 waves

// ---- workspace layout (bytes) ---------------------------------------------
static constexpr size_t OFF_FN    = 0;
static constexpr size_t OFF_MN    = OFF_FN    + (size_t)NROW * DIM * 2;
static constexpr size_t OFF_PN    = OFF_MN    + (size_t)NROW * DIM * 2;
static constexpr size_t OFF_MRAW  = OFF_PN    + (size_t)NROW * DIM * 2;
static constexpr size_t OFF_PRAW  = OFF_MRAW  + (size_t)NROW * DIM * 2;
static constexpr size_t OFF_W1BF  = OFF_PRAW  + (size_t)NROW * DIM * 2;
static constexpr size_t OFF_FMADJ = OFF_W1BF  + (size_t)DIM2 * DIM * 2;
static constexpr size_t OFF_FPADJ = OFF_FMADJ + (size_t)NROW * NROW * 2;
static constexpr size_t OFF_FEATS = OFF_FPADJ + (size_t)NROW * NROW * 2;
static constexpr size_t OFF_H     = OFF_FEATS + (size_t)NROW * DIM2 * 2;
static constexpr size_t OFF_MSKFM = OFF_H     + (size_t)NROW * DIM * 4;
static constexpr size_t OFF_MSKFP = OFF_MSKFM + (size_t)NROW * NROW * 4;
static constexpr size_t OFF_DEGFM = OFF_MSKFP + (size_t)NROW * NROW * 4;
static constexpr size_t OFF_DEGFP = OFF_DEGFM + (size_t)NROW * 4;
static constexpr size_t OFF_RSFM  = OFF_DEGFP + (size_t)NROW * 4;
static constexpr size_t OFF_RSFP  = OFF_RSFM  + (size_t)NROW * 4;
static constexpr size_t OFF_ACC   = OFF_RSFP  + (size_t)NROW * 4;   // posFM,negFM,posFP,negFP
static constexpr size_t OFF_ALLFM = OFF_ACC   + (size_t)4 * NROW * 4;
static constexpr size_t OFF_ALLFP = OFF_ALLFM + (size_t)NROW * 4;

// ---- async Global->LDS helpers (CDNA5) -------------------------------------
// Generic pointers into __shared__ carry the LDS byte offset in their low 32
// bits (flat-aperture rule), which is exactly what the async-load VDST wants.
__device__ __forceinline__ void async_load_b128(const void* gptr, void* lds_ptr) {
    unsigned lds = (unsigned)(uintptr_t)lds_ptr;
    unsigned long long g = (unsigned long long)(uintptr_t)gptr;
    asm volatile("global_load_async_to_lds_b128 %0, %1, off" :: "v"(lds), "v"(g) : "memory");
}
__device__ __forceinline__ void wait_async_le4() {
    asm volatile("s_wait_asynccnt 0x4" ::: "memory");
}
__device__ __forceinline__ void wait_async_0() {
    asm volatile("s_wait_asynccnt 0x0" ::: "memory");
}

// ---- WMMA fragment loaders --------------------------------------------------
// A fragment: 16x32 bf16 tile in LDS (row-major, stride ld).
__device__ __forceinline__ v16bf load_frag_a(const __bf16* lds, int ld, int lane) {
    int row  = lane & 15;
    int half = lane >> 4;
    v16bf f;
#pragma unroll
    for (int e = 0; e < 16; ++e) {
        int k = half * 16 + e;
        f[e] = lds[row * ld + k];
    }
    return f;
}

// B fragment: 32x16 bf16 tile in LDS laid out [k][n] (stride ld).
__device__ __forceinline__ v16bf load_frag_b(const __bf16* lds, int ld, int lane) {
    int col  = lane & 15;
    int half = lane >> 4;
    v16bf f;
#pragma unroll
    for (int e = 0; e < 16; ++e)
        f[e] = lds[(half * 16 + e) * ld + col];
    return f;
}

// B fragment from a transposed LDS tile laid out [n][k] (stride ld).
__device__ __forceinline__ v16bf load_frag_bt(const __bf16* lds, int ld, int lane) {
    int col  = lane & 15;
    int half = lane >> 4;
    v16bf f;
#pragma unroll
    for (int e = 0; e < 16; ++e)
        f[e] = lds[col * ld + half * 16 + e];
    return f;
}

// ---- small prep kernels ----------------------------------------------------
__global__ void convert_f32_bf16(const float* __restrict__ in, __bf16* __restrict__ out, int n) {
    for (int i = blockIdx.x * blockDim.x + threadIdx.x; i < n; i += gridDim.x * blockDim.x)
        out[i] = (__bf16)in[i];
}

__global__ void zero_f32(float* __restrict__ p, int n) {
    for (int i = blockIdx.x * blockDim.x + threadIdx.x; i < n; i += gridDim.x * blockDim.x)
        p[i] = 0.0f;
}

__global__ void normalize_kernel(const float* __restrict__ E, __bf16* __restrict__ out, int d) {
    __shared__ float red[TPB];
    int r = blockIdx.x, t = threadIdx.x;
    const float* e = E + (size_t)r * d;
    float ss = 0.f;
    for (int i = t; i < d; i += TPB) { float v = e[i]; ss += v * v; }
    red[t] = ss; __syncthreads();
    for (int s = TPB / 2; s > 0; s >>= 1) { if (t < s) red[t] += red[t + s]; __syncthreads(); }
    float inv = 1.f / fmaxf(sqrtf(red[0]), 1e-12f);
    for (int i = t; i < d; i += TPB) out[(size_t)r * d + i] = (__bf16)(e[i] * inv);
}

__global__ void adj_prep(const float* __restrict__ adj, __bf16* __restrict__ adjbf,
                         float* __restrict__ deg, float* __restrict__ rowscale, int n) {
    __shared__ float red[TPB];
    int r = blockIdx.x, t = threadIdx.x;
    const float* a = adj + (size_t)r * n;
    float s = 0.f;
    for (int i = t; i < n; i += TPB) { float v = a[i]; s += v; adjbf[(size_t)r * n + i] = (__bf16)v; }
    red[t] = s; __syncthreads();
    for (int k = TPB / 2; k > 0; k >>= 1) { if (t < k) red[t] += red[t + k]; __syncthreads(); }
    if (t == 0) { deg[r] = red[0]; rowscale[r] = 1.f / fmaxf(red[0], 1.f); }
}

// ---- tiled bf16 WMMA GEMM: C = A[MxK] @ B[KxN] -----------------------------
// 128x128 block tile, 8 waves (2x4), each wave 64x32 = 4x2 WMMA accumulators.
// Double-buffered LDS, async Global->LDS staging.
// MODE 1: out_bf[m][colOff+n] = acc * rowscale[m]   (aggregation -> feats)
// MODE 2: out_f [m][n]        = relu(acc + bias[n]) (MLP hidden layer)
template <int MODE>
__global__ void gemm_bf16(const __bf16* __restrict__ A, const __bf16* __restrict__ B,
                          int M, int N, int K,
                          float* __restrict__ outF, __bf16* __restrict__ outBF,
                          const float* __restrict__ rowscale, const float* __restrict__ bias,
                          int ldout, int colOff) {
    __shared__ __align__(16) __bf16 As[2][128 * 32];
    __shared__ __align__(16) __bf16 Bs[2][32 * 128];
    int t = threadIdx.x, lane = t & 31, wid = t >> 5;
    int waveM = wid >> 2, waveN = wid & 3;
    int m0 = blockIdx.y * 128, n0 = blockIdx.x * 128;

    // issue one K-tile worth of async loads: 2 A-chunks + 2 B-chunks / thread
    auto issue_tile = [&](int k0, int buf) {
#pragma unroll
        for (int u = 0; u < 2; ++u) {
            int c = t + u * TPB;
            int ar = c >> 2, ac = (c & 3) * 8;              // A: 128 rows x 32 k
            async_load_b128(&A[(size_t)(m0 + ar) * K + k0 + ac], &As[buf][ar * 32 + ac]);
        }
#pragma unroll
        for (int u = 0; u < 2; ++u) {
            int c = t + u * TPB;
            int br = c >> 4, bc = (c & 15) * 8;             // B: 32 k x 128 cols
            async_load_b128(&B[(size_t)(k0 + br) * N + n0 + bc], &Bs[buf][br * 128 + bc]);
        }
    };

    v8f zero = {};
    v8f acc[4][2];
#pragma unroll
    for (int i = 0; i < 4; ++i)
#pragma unroll
        for (int j = 0; j < 2; ++j) acc[i][j] = zero;

    issue_tile(0, 0);
    int buf = 0;
    for (int k0 = 0; k0 < K; k0 += 32, buf ^= 1) {
        if (k0 + 32 < K) {
            issue_tile(k0 + 32, buf ^ 1);   // prefetch next tile into other buffer
            wait_async_le4();               // current tile (4 older ops) has landed
        } else {
            wait_async_0();
        }
        __syncthreads();

        v16bf af[4], bq[2];
#pragma unroll
        for (int mt = 0; mt < 4; ++mt)
            af[mt] = load_frag_a(&As[buf][(waveM * 64 + mt * 16) * 32], 32, lane);
#pragma unroll
        for (int nt = 0; nt < 2; ++nt)
            bq[nt] = load_frag_b(&Bs[buf][waveN * 32 + nt * 16], 128, lane);
#pragma unroll
        for (int mt = 0; mt < 4; ++mt)
#pragma unroll
            for (int nt = 0; nt < 2; ++nt)
                acc[mt][nt] = __builtin_amdgcn_wmma_f32_16x16x32_bf16(
                    false, af[mt], false, bq[nt], (short)0, acc[mt][nt], false, false);
        __syncthreads();   // everyone done reading buf before it is overwritten
    }

    int half = lane >> 4, nn = lane & 15;
#pragma unroll
    for (int mt = 0; mt < 4; ++mt)
#pragma unroll
        for (int nt = 0; nt < 2; ++nt)
#pragma unroll
            for (int r = 0; r < 8; ++r) {
                int m = m0 + waveM * 64 + mt * 16 + r + 8 * half;
                int n = n0 + waveN * 32 + nt * 16 + nn;
                float v = acc[mt][nt][r];
                if (MODE == 1)
                    outBF[(size_t)m * ldout + colOff + n] = (__bf16)(v * rowscale[m]);
                else
                    outF[(size_t)m * ldout + n] = fmaxf(v + bias[n], 0.f);
            }
}

// ---- similarity GEMM: S = Fn @ En^T with fused exp/mask/row-sum epilogue ---
__global__ void sim_kernel(const __bf16* __restrict__ Fn, const __bf16* __restrict__ En,
                           const float* __restrict__ adj, float* __restrict__ maskedS,
                           float* __restrict__ pos, float* __restrict__ negsum, int K) {
    __shared__ __align__(16) __bf16 As[2][128 * 32];
    __shared__ __align__(16) __bf16 Bt[2][128 * 32];   // [n][k]
    __shared__ float posRow[128], negRow[128];
    int t = threadIdx.x, lane = t & 31, wid = t >> 5;
    int waveM = wid >> 2, waveN = wid & 3;
    int m0 = blockIdx.y * 128, n0 = blockIdx.x * 128;
    if (t < 128) { posRow[t] = 0.f; negRow[t] = 0.f; }

    auto issue_tile = [&](int k0, int buf) {
#pragma unroll
        for (int u = 0; u < 2; ++u) {
            int c = t + u * TPB;
            int ar = c >> 2, ac = (c & 3) * 8;
            async_load_b128(&Fn[(size_t)(m0 + ar) * K + k0 + ac], &As[buf][ar * 32 + ac]);
        }
#pragma unroll
        for (int u = 0; u < 2; ++u) {
            int c = t + u * TPB;
            int nr = c >> 2, kc = (c & 3) * 8;
            async_load_b128(&En[(size_t)(n0 + nr) * K + k0 + kc], &Bt[buf][nr * 32 + kc]);
        }
    };

    v8f zero = {};
    v8f acc[4][2];
#pragma unroll
    for (int i = 0; i < 4; ++i)
#pragma unroll
        for (int j = 0; j < 2; ++j) acc[i][j] = zero;

    issue_tile(0, 0);
    int buf = 0;
    for (int k0 = 0; k0 < K; k0 += 32, buf ^= 1) {
        if (k0 + 32 < K) {
            issue_tile(k0 + 32, buf ^ 1);
            wait_async_le4();
        } else {
            wait_async_0();
        }
        __syncthreads();

        v16bf af[4], bq[2];
#pragma unroll
        for (int mt = 0; mt < 4; ++mt)
            af[mt] = load_frag_a(&As[buf][(waveM * 64 + mt * 16) * 32], 32, lane);
#pragma unroll
        for (int nt = 0; nt < 2; ++nt)
            bq[nt] = load_frag_bt(&Bt[buf][(waveN * 32 + nt * 16) * 32], 32, lane);
#pragma unroll
        for (int mt = 0; mt < 4; ++mt)
#pragma unroll
            for (int nt = 0; nt < 2; ++nt)
                acc[mt][nt] = __builtin_amdgcn_wmma_f32_16x16x32_bf16(
                    false, af[mt], false, bq[nt], (short)0, acc[mt][nt], false, false);
        __syncthreads();
    }

    int half = lane >> 4, nn = lane & 15;
#pragma unroll
    for (int mt = 0; mt < 4; ++mt)
#pragma unroll
        for (int nt = 0; nt < 2; ++nt)
#pragma unroll
            for (int r = 0; r < 8; ++r) {
                int mloc = waveM * 64 + mt * 16 + r + 8 * half;
                int m = m0 + mloc;
                int n = n0 + waveN * 32 + nt * 16 + nn;
                float s = acc[mt][nt][r];
                float adjv = adj[(size_t)m * NROW + n];
                float sim = __expf(s * 10.0f);                 // 1/TAU = 10
                bool ispos = adjv > 0.f;
                maskedS[(size_t)m * NROW + n] = ispos ? -2.0f : s;
                float pc = ispos ? sim * adjv : 0.f;
                float nc = ispos ? 0.f : sim;
#pragma unroll
                for (int o = 8; o >= 1; o >>= 1) {
                    pc += __shfl_xor(pc, o, 16);
                    nc += __shfl_xor(nc, o, 16);
                }
                if (nn == 0) { atomicAdd(&posRow[mloc], pc); atomicAdd(&negRow[mloc], nc); }
            }
    __syncthreads();
    if (t < 128) {
        atomicAdd(&pos[m0 + t], posRow[t]);
        atomicAdd(&negsum[m0 + t], negRow[t]);
    }
}

// ---- per-row hard-negative top-k via binary search on masked cosine --------
__global__ void topk_kernel(const float* __restrict__ masked, const float* __restrict__ pos,
                            const float* __restrict__ negsum, float* __restrict__ allOut) {
    __shared__ float row[NROW];
    __shared__ float red[TPB];
    __shared__ int   ired[TPB];
    int r = blockIdx.x, t = threadIdx.x;
    const float* src = masked + (size_t)r * NROW;
    for (int i = t; i < NROW; i += TPB) row[i] = src[i];
    __syncthreads();

    int cnt = 0;
    for (int i = t; i < NROW; i += TPB) if (row[i] > -1.5f) cnt++;
    ired[t] = cnt; __syncthreads();
    for (int s = TPB / 2; s > 0; s >>= 1) { if (t < s) ired[t] += ired[t + s]; __syncthreads(); }
    int num_neg = ired[0]; __syncthreads();

    int k = 0;
    if (num_neg > 0) {
        k = (int)(num_neg * 0.3f);
        if (k < 1) k = 1;
        if (k > num_neg) k = num_neg;
    }
    float lo = -1.001f, hi = 1.001f;
    for (int it = 0; it < 24; ++it) {
        float mid = 0.5f * (lo + hi);
        int c = 0;
        for (int i = t; i < NROW; i += TPB) if (row[i] > mid) c++;
        ired[t] = c; __syncthreads();
        for (int s = TPB / 2; s > 0; s >>= 1) { if (t < s) ired[t] += ired[t + s]; __syncthreads(); }
        int ctot = ired[0]; __syncthreads();
        if (ctot > k) lo = mid; else hi = mid;
    }
    float thr = lo, es = 0.f;
    for (int i = t; i < NROW; i += TPB) { float v = row[i]; if (v > thr) es += __expf(v * 10.0f); }
    red[t] = es; __syncthreads();
    for (int s = TPB / 2; s > 0; s >>= 1) { if (t < s) red[t] += red[t + s]; __syncthreads(); }
    if (t == 0) allOut[r] = pos[r] + negsum[r] + red[0];
}

// ---- attention weights: logits = h @ W2 + b2, double softmax ----------------
__global__ void weights_kernel(const float* __restrict__ h, const float* __restrict__ W2,
                               const float* __restrict__ b2, float* __restrict__ outw) {
    __shared__ float r0[TPB], r1[TPB];
    int r = blockIdx.x, t = threadIdx.x;
    const float* hr = h + (size_t)r * DIM;   // already relu'd
    float a0 = 0.f, a1 = 0.f;
    for (int i = t; i < DIM; i += TPB) {
        float v = hr[i];
        a0 += v * W2[i * 2 + 0];
        a1 += v * W2[i * 2 + 1];
    }
    r0[t] = a0; r1[t] = a1; __syncthreads();
    for (int s = TPB / 2; s > 0; s >>= 1) {
        if (t < s) { r0[t] += r0[t + s]; r1[t] += r1[t + s]; }
        __syncthreads();
    }
    if (t == 0) {
        float l0 = r0[0] + b2[0], l1 = r1[0] + b2[1];
        float m = fmaxf(l0, l1);
        float e0 = __expf(l0 - m), e1 = __expf(l1 - m), s = e0 + e1;
        float w0 = e0 / s, w1 = e1 / s;
        float t0 = w0 / 0.7f, t1 = w1 / 0.7f;            // WEIGHT_TEMP
        m = fmaxf(t0, t1);
        e0 = __expf(t0 - m); e1 = __expf(t1 - m); s = e0 + e1;
        outw[r * 2 + 0] = e0 / s;
        outw[r * 2 + 1] = e1 / s;
    }
}

// ---- final loss reduction ---------------------------------------------------
__global__ void loss_kernel(const float* __restrict__ posFM, const float* __restrict__ posFP,
                            const float* __restrict__ allFM, const float* __restrict__ allFP,
                            const float* __restrict__ degFM, const float* __restrict__ degFP,
                            const float* __restrict__ w, float* __restrict__ out) {
    __shared__ float red[TPB];
    int t = threadIdx.x;
    float acc = 0.f;
    for (int r = blockIdx.x * blockDim.x + t; r < NROW; r += gridDim.x * blockDim.x) {
        float w0 = w[r * 2], w1 = w[r * 2 + 1];
        float wp  = w0 * posFM[r] + w1 * posFP[r];
        float den = w0 * allFM[r] + w1 * allFP[r];
        float nei = fmaxf(degFM[r] + degFP[r], 1.f);
        float loss = wp / fmaxf(den, 1e-10f);
        loss = fmaxf(loss / nei, 1e-10f);
        acc += -__logf(loss);
    }
    red[t] = acc; __syncthreads();
    for (int s = TPB / 2; s > 0; s >>= 1) { if (t < s) red[t] += red[t + s]; __syncthreads(); }
    if (t == 0) atomicAdd(out, red[0] / (float)NROW);
}

// ---------------------------------------------------------------------------
extern "C" void kernel_launch(void* const* d_in, const int* in_sizes, int n_in,
                              void* d_out, int out_size, void* d_ws, size_t ws_size,
                              hipStream_t stream) {
    const float* embF  = (const float*)d_in[0];
    const float* embM  = (const float*)d_in[1];
    const float* embP  = (const float*)d_in[2];
    const float* FMadj = (const float*)d_in[3];
    const float* FPadj = (const float*)d_in[4];
    const float* W1    = (const float*)d_in[5];
    const float* b1    = (const float*)d_in[6];
    const float* W2    = (const float*)d_in[7];
    const float* b2    = (const float*)d_in[8];
    float* out = (float*)d_out;

    char* ws = (char*)d_ws;
    __bf16* Fn     = (__bf16*)(ws + OFF_FN);
    __bf16* Mn     = (__bf16*)(ws + OFF_MN);
    __bf16* Pn     = (__bf16*)(ws + OFF_PN);
    __bf16* Mraw   = (__bf16*)(ws + OFF_MRAW);
    __bf16* Praw   = (__bf16*)(ws + OFF_PRAW);
    __bf16* W1bf   = (__bf16*)(ws + OFF_W1BF);
    __bf16* FMbf   = (__bf16*)(ws + OFF_FMADJ);
    __bf16* FPbf   = (__bf16*)(ws + OFF_FPADJ);
    __bf16* feats  = (__bf16*)(ws + OFF_FEATS);
    float*  h      = (float*)(ws + OFF_H);
    float*  mskFM  = (float*)(ws + OFF_MSKFM);
    float*  mskFP  = (float*)(ws + OFF_MSKFP);
    float*  degFM  = (float*)(ws + OFF_DEGFM);
    float*  degFP  = (float*)(ws + OFF_DEGFP);
    float*  rsFM   = (float*)(ws + OFF_RSFM);
    float*  rsFP   = (float*)(ws + OFF_RSFP);
    float*  posFM  = (float*)(ws + OFF_ACC);
    float*  negFM  = posFM + NROW;
    float*  posFP  = negFM + NROW;
    float*  negFP  = posFP + NROW;
    float*  allFM  = (float*)(ws + OFF_ALLFM);
    float*  allFP  = (float*)(ws + OFF_ALLFP);

    // 1) precision conversions
    convert_f32_bf16<<<512, TPB, 0, stream>>>(embM, Mraw, NROW * DIM);
    convert_f32_bf16<<<512, TPB, 0, stream>>>(embP, Praw, NROW * DIM);
    convert_f32_bf16<<<256, TPB, 0, stream>>>(W1, W1bf, DIM2 * DIM);

    // 2) row-normalized bf16 embeddings
    normalize_kernel<<<NROW, TPB, 0, stream>>>(embF, Fn, DIM);
    normalize_kernel<<<NROW, TPB, 0, stream>>>(embM, Mn, DIM);
    normalize_kernel<<<NROW, TPB, 0, stream>>>(embP, Pn, DIM);

    // 3) adjacency prep (bf16 copy + degree + 1/deg)
    adj_prep<<<NROW, TPB, 0, stream>>>(FMadj, FMbf, degFM, rsFM, NROW);
    adj_prep<<<NROW, TPB, 0, stream>>>(FPadj, FPbf, degFP, rsFP, NROW);

    // 4) zero accumulators + loss scalar
    zero_f32<<<64, TPB, 0, stream>>>(posFM, 4 * NROW);
    zero_f32<<<1, 32, 0, stream>>>(out, 1);

    // 5) aggregation GEMMs (WMMA bf16), fused /deg and bf16 store into feats
    dim3 gAgg(DIM / 128, NROW / 128);
    gemm_bf16<1><<<gAgg, TPB, 0, stream>>>(FMbf, Mraw, NROW, DIM, NROW,
                                           nullptr, feats, rsFM, nullptr, DIM2, 0);
    gemm_bf16<1><<<gAgg, TPB, 0, stream>>>(FPbf, Praw, NROW, DIM, NROW,
                                           nullptr, feats, rsFP, nullptr, DIM2, DIM);

    // 6) MLP hidden layer (WMMA bf16), fused bias + relu
    gemm_bf16<2><<<gAgg, TPB, 0, stream>>>(feats, W1bf, NROW, DIM, DIM2,
                                           h, nullptr, nullptr, b1, DIM, 0);

    // 7) similarity GEMMs (WMMA bf16) with fused exp/mask/pos-neg row sums
    dim3 gSim(NROW / 128, NROW / 128);
    sim_kernel<<<gSim, TPB, 0, stream>>>(Fn, Mn, FMadj, mskFM, posFM, negFM, DIM);
    sim_kernel<<<gSim, TPB, 0, stream>>>(Fn, Pn, FPadj, mskFP, posFP, negFP, DIM);

    // 8) per-row hard-negative top-k sums
    topk_kernel<<<NROW, TPB, 0, stream>>>(mskFM, posFM, negFM, allFM);
    topk_kernel<<<NROW, TPB, 0, stream>>>(mskFP, posFP, negFP, allFP);

    // 9) attention weights -> d_out[1..8192]
    weights_kernel<<<NROW, TPB, 0, stream>>>(h, W2, b2, out + 1);

    // 10) scalar loss -> d_out[0]
    loss_kernel<<<64, TPB, 0, stream>>>(posFM, posFP, allFM, allFP,
                                        degFM, degFP, out + 1, out);
}
